// interframe_att_57243324121094
// MI455X (gfx1250) — matrix operand order repaired and checked by reading, not compiled, not run on hardware
//
#include <hip/hip_runtime.h>
#include <hip/hip_bf16.h>
#include <stdint.h>

// ---------------------------------------------------------------------------
// Mamba block forward on gfx1250 (wave32, WMMA).
//   B=16, L=1024, D_MODEL=256, D_INNER=512, D_STATE=16, D_CONV=4, DT_RANK=16
// Pipeline:
//   1. f32->f16 converts (token, in_proj_w, x_proj_w, out_proj_w)
//   2. WMMA GEMM (2x2 blocked)  xz = token @ in_proj_w^T   (16384x1024, K=256)
//   3. depthwise causal conv + SiLU -> xact (f32 + f16)
//   4. WMMA GEMM (1x1)  dbc = xact @ x_proj_w^T            (16384x48,   K=512)
//   5. zero-pad dt (K 16->32) + dt_proj_w
//   6. WMMA GEMM (2x2)  delta_raw = dt @ dt_proj_w^T       (16384x512,  K=32)
//   7. selective scan fused with skip + SiLU(z) gate -> f16
//   8. WMMA GEMM (2x2)  out = y @ out_proj_w^T             (16384x256,  K=512)
// ---------------------------------------------------------------------------

#define BT     16
#define LSEQ   1024
#define DMODEL 256
#define DINNER 512
#define DSTATE 16
#define DCONV  4
#define DTRANK 16
#define MROWS  (BT * LSEQ)           // 16384
#define DBCW   (DTRANK + 2 * DSTATE) // 48

typedef __attribute__((ext_vector_type(16))) _Float16 v16h;
typedef __attribute__((ext_vector_type(8)))  _Float16 v8h;
typedef __attribute__((ext_vector_type(8)))  float    v8f;

// ---------------------------------------------------------------- converts
__global__ void k_f32_to_f16(const float* __restrict__ src,
                             _Float16* __restrict__ dst, int n) {
  int i = blockIdx.x * blockDim.x + threadIdx.x;
  if (i < n) dst[i] = (_Float16)src[i];
}

// pad/convert: dst[r, c<src_cols] = src[r*src_ld + c], else 0
__global__ void k_pad_f16(const float* __restrict__ src,
                          _Float16* __restrict__ dst,
                          int rows, int src_ld, int src_cols, int dst_ld) {
  int i = blockIdx.x * blockDim.x + threadIdx.x;
  if (i >= rows * dst_ld) return;
  int r = i / dst_ld, c = i - r * dst_ld;
  dst[i] = (c < src_cols) ? (_Float16)src[r * src_ld + c] : (_Float16)0.0f;
}

// ---------------------------------------------------------- fragment loaders
// A fragment (16-bit A 16x32): per lane, elements 0..15 = K hi*16+0..15
// relative to k0 (matches B).  `row` must already include K stride.
__device__ __forceinline__ v16h load_afrag(const _Float16* row, int k0, int hi) {
  return *(const v16h*)(row + k0 + hi * 16);
}
// B fragment: lane n holds B[k0 + hi*16 .. +15, n] (contiguous in weight row)
__device__ __forceinline__ v16h load_bfrag(const _Float16* row, int k0, int hi) {
  return *(const v16h*)(row + k0 + hi * 16);
}
__device__ __forceinline__ v8f wmma_f16(v16h a, v16h b, v8f c) {
  return __builtin_amdgcn_wmma_f32_16x16x32_f16(false, a, false, b,
                                                (short)0, c, false, false);
}

// ------------------------------------------------- WMMA GEMM, 1x1 tile/wave
// C[M,N] f32 row-major = A[M,K] f16 row-major * B[N,K]^T (weights row = K taps)
// K%32==0, M%16==0, N%16==0.
__global__ void k_wmma_gemm(const _Float16* __restrict__ A,
                            const _Float16* __restrict__ B,
                            float* __restrict__ C, int M, int N, int K) {
  const int lane = threadIdx.x & 31;
  const int wid  = (blockIdx.x * blockDim.x + threadIdx.x) >> 5;
  const int tiles_n = N >> 4;
  if (wid >= (M >> 4) * tiles_n) return;     // wave-uniform: EXEC stays all-1s
  const int tm = (wid / tiles_n) << 4;
  const int tn = (wid % tiles_n) << 4;
  const int hi = lane >> 4, l16 = lane & 15;
  const _Float16* arow = A + (size_t)(tm + l16) * K;
  const _Float16* brow = B + (size_t)(tn + l16) * K;
  v8f acc = {};
  for (int k0 = 0; k0 < K; k0 += 32)
    acc = wmma_f16(load_afrag(arow, k0, hi), load_bfrag(brow, k0, hi), acc);
  float* crow = C + (size_t)(tm + hi * 8) * N + tn + l16;
#pragma unroll
  for (int r = 0; r < 8; ++r) crow[(size_t)r * N] = acc[r];
}

// ------------------------------------------------- WMMA GEMM, 2x2 tiles/wave
// 32x32 C per wave: A/B fragments each reused by two WMMAs (2x arithmetic
// intensity vs 1x1).  Requires M%32==0, N%32==0, K%32==0.
__global__ void k_wmma_gemm2x2(const _Float16* __restrict__ A,
                               const _Float16* __restrict__ B,
                               float* __restrict__ C, int M, int N, int K) {
  const int lane = threadIdx.x & 31;
  const int wid  = (blockIdx.x * blockDim.x + threadIdx.x) >> 5;
  const int tiles_n = N >> 5;
  if (wid >= (M >> 5) * tiles_n) return;     // wave-uniform
  const int tm = (wid / tiles_n) << 5;
  const int tn = (wid % tiles_n) << 5;
  const int hi = lane >> 4, l16 = lane & 15;
  const _Float16* arow0 = A + (size_t)(tm + l16) * K;
  const _Float16* arow1 = arow0 + (size_t)16 * K;
  const _Float16* brow0 = B + (size_t)(tn + l16) * K;
  const _Float16* brow1 = brow0 + (size_t)16 * K;
  v8f acc00 = {}, acc01 = {}, acc10 = {}, acc11 = {};
  for (int k0 = 0; k0 < K; k0 += 32) {
    v16h a0 = load_afrag(arow0, k0, hi);
    v16h a1 = load_afrag(arow1, k0, hi);
    v16h b0 = load_bfrag(brow0, k0, hi);
    v16h b1 = load_bfrag(brow1, k0, hi);
    acc00 = wmma_f16(a0, b0, acc00);
    acc01 = wmma_f16(a0, b1, acc01);
    acc10 = wmma_f16(a1, b0, acc10);
    acc11 = wmma_f16(a1, b1, acc11);
  }
  float* c00 = C + (size_t)(tm + hi * 8) * N + tn + l16;
#pragma unroll
  for (int r = 0; r < 8; ++r) {
    size_t ro = (size_t)r * N;
    c00[ro]                    = acc00[r];
    c00[ro + 16]               = acc01[r];
    c00[ro + (size_t)16 * N]      = acc10[r];
    c00[ro + (size_t)16 * N + 16] = acc11[r];
  }
}

// ------------------------------------------------- depthwise conv1d + SiLU
__global__ void k_conv_silu(const float* __restrict__ xz,      // MROWS x 1024
                            const float* __restrict__ conv_w,  // 512 x 4
                            const float* __restrict__ conv_b,  // 512
                            float* __restrict__ xact,          // MROWS x 512
                            _Float16* __restrict__ xacth) {    // MROWS x 512
  int i = blockIdx.x * blockDim.x + threadIdx.x;
  if (i >= MROWS * DINNER) return;
  int d = i & (DINNER - 1);
  int m = i >> 9;
  int t = m & (LSEQ - 1);
  int mb = m - t;
  float acc = conv_b[d];
#pragma unroll
  for (int k = 0; k < DCONV; ++k) {
    int tt = t + k - (DCONV - 1);
    if (tt >= 0)
      acc += xz[(size_t)(mb + tt) * (2 * DINNER) + d] * conv_w[d * DCONV + k];
  }
  float s = acc / (1.0f + __expf(-acc));
  xact[i]  = s;
  xacth[i] = (_Float16)s;
}

// --------------------------------------- selective scan + skip + gate (fused)
// wave = (batch b, channel pair). lane = dsub*16 + n : 2 channels x 16 states.
// h lives in a register; y = <h,C> reduced across the 16 state lanes via
// shfl_xor (masks 1/2/4/8 stay inside each 16-lane group under wave32).
// Lane n==0 applies the D_skip residual and SiLU(z) gate and writes f16.
__global__ void k_scan_gate(const float* __restrict__ delta_raw, // MROWS x 512
                            const float* __restrict__ dbc,       // MROWS x 48
                            const float* __restrict__ xact,      // MROWS x 512
                            const float* __restrict__ xz,        // MROWS x 1024
                            const float* __restrict__ A_log,     // 512 x 16
                            const float* __restrict__ dt_bias,   // 512
                            const float* __restrict__ Dskip,     // 512
                            _Float16* __restrict__ ygh) {        // MROWS x 512
  const int lane = threadIdx.x & 31;
  const int wid  = (blockIdx.x * blockDim.x + threadIdx.x) >> 5;
  if (wid >= BT * (DINNER / 2)) return;
  const int b     = wid >> 8;
  const int dpair = wid & 255;
  const int dsub  = lane >> 4;
  const int n     = lane & 15;
  const int d     = dpair * 2 + dsub;

  const float Adn  = -__expf(A_log[d * DSTATE + n]);   // A = -exp(A_log)
  const float bias = dt_bias[d];
  const float Dsk  = Dskip[d];
  const int   m0   = b << 10;

  float h = 0.0f;
  for (int t = 0; t < LSEQ; ++t) {
    const int m = m0 + t;
    if (t + 8 < LSEQ) __builtin_prefetch(dbc + (size_t)(m + 8) * DBCW, 0, 0);
    float draw  = delta_raw[(size_t)m * DINNER + d] + bias;
    float delta = (draw > 20.0f) ? draw : log1pf(__expf(draw)); // softplus
    float Bn = dbc[(size_t)m * DBCW + DTRANK + n];
    float Cn = dbc[(size_t)m * DBCW + DTRANK + DSTATE + n];
    float xv = xact[(size_t)m * DINNER + d];
    float dA = __expf(delta * Adn);
    h = dA * h + delta * Bn * xv;                      // recurrence
    float p = h * Cn;                                  // <h, C> over n
    p += __shfl_xor(p, 1, 32);
    p += __shfl_xor(p, 2, 32);
    p += __shfl_xor(p, 4, 32);
    p += __shfl_xor(p, 8, 32);
    if (n == 0) {
      float z = xz[(size_t)m * (2 * DINNER) + DINNER + d];
      float y = p + xv * Dsk;
      float g = y * (z / (1.0f + __expf(-z)));         // y * silu(z)
      ygh[(size_t)m * DINNER + d] = (_Float16)g;
    }
  }
}

// ---------------------------------------------------------------------------
extern "C" void kernel_launch(void* const* d_in, const int* in_sizes, int n_in,
                              void* d_out, int out_size, void* d_ws,
                              size_t ws_size, hipStream_t stream) {
  const float* token     = (const float*)d_in[0];  // 16x1024x256
  const float* in_proj_w = (const float*)d_in[1];  // 1024x256
  const float* conv_w    = (const float*)d_in[2];  // 512x4
  const float* conv_b    = (const float*)d_in[3];  // 512
  const float* x_proj_w  = (const float*)d_in[4];  // 48x512
  const float* dt_proj_w = (const float*)d_in[5];  // 512x16
  const float* dt_proj_b = (const float*)d_in[6];  // 512
  const float* A_log     = (const float*)d_in[7];  // 512x16
  const float* D_skip    = (const float*)d_in[8];  // 512
  const float* out_proj_w= (const float*)d_in[9];  // 256x512
  float* out = (float*)d_out;                      // 16x1024x256

  // ---- workspace layout ----
  char* ws = (char*)d_ws;
  size_t off = 0;
  auto alloc = [&](size_t bytes) { char* p = ws + off; off += (bytes + 255) & ~(size_t)255; return p; };
  _Float16* tokh   = (_Float16*)alloc((size_t)MROWS * DMODEL * 2);     // 8.0 MB
  _Float16* w_inh  = (_Float16*)alloc((size_t)(2 * DINNER) * DMODEL * 2);
  _Float16* w_xph  = (_Float16*)alloc((size_t)DBCW * DINNER * 2);
  _Float16* w_dth  = (_Float16*)alloc((size_t)DINNER * 32 * 2);
  _Float16* w_outh = (_Float16*)alloc((size_t)DMODEL * DINNER * 2);
  float*    xz     = (float*)   alloc((size_t)MROWS * 2 * DINNER * 4); // 67 MB
  float*    xact   = (float*)   alloc((size_t)MROWS * DINNER * 4);     // 33.5 MB
  _Float16* xacth  = (_Float16*)alloc((size_t)MROWS * DINNER * 2);     // 16.8 MB
  float*    dbc    = (float*)   alloc((size_t)MROWS * DBCW * 4);       // 3.1 MB
  _Float16* dt16   = (_Float16*)alloc((size_t)MROWS * 32 * 2);         // 1.0 MB
  float*    draw   = (float*)   alloc((size_t)MROWS * DINNER * 4);     // 33.5 MB
  _Float16* ygh    = (_Float16*)alloc((size_t)MROWS * DINNER * 2);     // 16.8 MB
  (void)ws_size; (void)n_in; (void)in_sizes; (void)out_size;

  const int EB = 256;
  auto eg = [](int n, int b) { return (n + b - 1) / b; };

  // 1. f32 -> f16 converts
  k_f32_to_f16<<<eg(MROWS * DMODEL, EB), EB, 0, stream>>>(token, tokh, MROWS * DMODEL);
  k_f32_to_f16<<<eg(2 * DINNER * DMODEL, EB), EB, 0, stream>>>(in_proj_w, w_inh, 2 * DINNER * DMODEL);
  k_f32_to_f16<<<eg(DBCW * DINNER, EB), EB, 0, stream>>>(x_proj_w, w_xph, DBCW * DINNER);
  k_f32_to_f16<<<eg(DMODEL * DINNER, EB), EB, 0, stream>>>(out_proj_w, w_outh, DMODEL * DINNER);

  // 2. in_proj GEMM: xz[16384,1024] = tokh @ w_inh^T  (2x2 blocked)
  {
    int waves = (MROWS / 32) * (2 * DINNER / 32);      // 16384
    k_wmma_gemm2x2<<<waves / 4, 128, 0, stream>>>(tokh, w_inh, xz, MROWS, 2 * DINNER, DMODEL);
  }

  // 3. depthwise conv + SiLU
  k_conv_silu<<<eg(MROWS * DINNER, EB), EB, 0, stream>>>(xz, conv_w, conv_b, xact, xacth);

  // 4. x_proj GEMM: dbc[16384,48] = xacth @ w_xph^T   (N=48 -> 1x1 path)
  {
    int waves = (MROWS / 16) * (DBCW / 16);            // 3072
    k_wmma_gemm<<<waves / 4, 128, 0, stream>>>(xacth, w_xph, dbc, MROWS, DBCW, DINNER);
  }

  // 5. zero-pad dt (first 16 cols of dbc) and dt_proj_w to K=32
  k_pad_f16<<<eg(MROWS * 32, EB), EB, 0, stream>>>(dbc, dt16, MROWS, DBCW, DTRANK, 32);
  k_pad_f16<<<eg(DINNER * 32, EB), EB, 0, stream>>>(dt_proj_w, w_dth, DINNER, DTRANK, DTRANK, 32);

  // 6. dt_proj GEMM: draw[16384,512] = dt16 @ w_dth^T (2x2 blocked)
  {
    int waves = (MROWS / 32) * (DINNER / 32);          // 8192
    k_wmma_gemm2x2<<<waves / 4, 128, 0, stream>>>(dt16, w_dth, draw, MROWS, DINNER, 32);
  }

  // 7. selective scan fused with skip + gate: 4096 waves
  {
    int waves = BT * (DINNER / 2);                     // 4096
    k_scan_gate<<<waves * 32 / 256, 256, 0, stream>>>(draw, dbc, xact, xz, A_log,
                                                      dt_proj_b, D_skip, ygh);
  }

  // 8. out_proj GEMM: out[16384,256] = ygh @ w_outh^T (2x2 blocked)
  {
    int waves = (MROWS / 32) * (DMODEL / 32);          // 4096
    k_wmma_gemm2x2<<<waves / 4, 128, 0, stream>>>(ygh, w_outh, out, MROWS, DMODEL, DINNER);
  }
}
